// MPNN_1649267442269
// MI455X (gfx1250) — matrix operand order repaired and checked by reading, not compiled, not run on hardware
//
#include <hip/hip_runtime.h>
#include <stdint.h>

#define NN 32768      // nodes
#define NE 262144     // edges
#define NBATCH 8
#define TWIN 25
#define HD 128
#define NLAY 6

// LDS layout (bytes) for the 9-chunk GEMM kernels (edge / update):
//   [0, 73728)        swizzled W1 fragments (9 chunks * 8 KB)
//   [73728, 106496)   swizzled W2 fragments (4 chunks * 8 KB)
//   [106496, 139264)  per-wave 16x128 f16 activation transpose buffers
#define W1_BYTES_9   73728
#define W2_BYTES     32768
#define ACT_OFF_9    106496
#define SMEM_9       139264
// Encoder (1-chunk W1):
#define W1_BYTES_1   8192
#define ACT_OFF_1    40960
#define SMEM_1       73728

typedef _Float16 f16;
typedef __attribute__((ext_vector_type(16))) _Float16 v16h;
typedef __attribute__((ext_vector_type(8)))  float    v8f;

union AFrag { v16h v; uint4 q[2]; };

// A-fragment (16x32 f16) per CDNA5 ISA: lane L -> row L&15, kbase (L>>4)*8,
// dwords i<4: K = kbase+2i, dwords i>=4: K = 16+kbase+2(i-4)
// => two contiguous 16-byte segments at f16 offsets (k0+kb2) and (k0+kb2+16).
__device__ __forceinline__ v16h load_a_frag(const f16* prow, int k0, int kb2) {
  AFrag f;
  f.q[0] = *(const uint4*)(prow + k0 + kb2);
  f.q[1] = *(const uint4*)(prow + k0 + kb2 + 16);
  return f.v;
}

__device__ __forceinline__ float swish_f(float x) {
  return x / (1.0f + __expf(-x));
}

__device__ __forceinline__ void atomic_add_f32(float* p, float v) {
  __hip_atomic_fetch_add(p, v, __ATOMIC_RELAXED, __HIP_MEMORY_SCOPE_AGENT);
}

// ---------------------------------------------------------------------------
// Weight swizzle: W (Krows x 128, f32 row-major, zero-padded to nchunks*32 K)
// -> B fragments: frag(c,t): lane L holds col t*16+(L&15), K-base (L>>4)*8 (+8 for pairs i>=4),
// 16 f16 (8 packed dword pairs) per lane, fragment stride 512 f16.
// ---------------------------------------------------------------------------
__global__ void k_swizzle(const float* __restrict__ W, int Krows, int nchunks,
                          f16* __restrict__ dst) {
  int id = blockIdx.x * blockDim.x + threadIdx.x;
  int total = nchunks * 8 * 32;
  if (id >= total) return;
  int lane = id & 31;
  int t    = (id >> 5) & 7;
  int c    = id >> 8;
  int col  = t * 16 + (lane & 15);
  int kb   = (lane >> 4) * 8;
  f16* d = dst + (size_t)id * 16;
#pragma unroll
  for (int i = 0; i < 8; i++) {
    int k = c * 32 + kb + 2 * i + (i >= 4 ? 8 : 0);
    float v0 = (k     < Krows) ? W[(size_t)k * HD + col]       : 0.0f;
    float v1 = (k + 1 < Krows) ? W[(size_t)(k + 1) * HD + col] : 0.0f;
    d[2 * i]     = (f16)v0;
    d[2 * i + 1] = (f16)v1;
  }
}

__global__ void k_zero(float* __restrict__ p, int count) {
  int i = blockIdx.x * blockDim.x + threadIdx.x;
  if (i < count) p[i] = 0.0f;
}

// node features: [u(25), pos_x/16, pos_t/4, 0...] ; update feat: [pos_t/4, 0...]
__global__ void k_nodefeat(const float* __restrict__ u, const float* __restrict__ pos,
                           f16* __restrict__ nfeat, f16* __restrict__ upfeat) {
  int n = blockIdx.x * blockDim.x + threadIdx.x;
  if (n >= NN) return;
  float pt = pos[n * 2 + 0] * 0.25f;
  float px = pos[n * 2 + 1] * (1.0f / 16.0f);
  f16* d = nfeat + (size_t)n * 32;
#pragma unroll
  for (int k = 0; k < 25; k++) d[k] = (f16)u[(size_t)n * TWIN + k];
  d[25] = (f16)px;
  d[26] = (f16)pt;
#pragma unroll
  for (int k = 27; k < 32; k++) d[k] = (f16)0.0f;
  f16* e = upfeat + (size_t)n * 32;
  e[0] = (f16)pt;
#pragma unroll
  for (int k = 1; k < 32; k++) e[k] = (f16)0.0f;
}

// edge features: [u_diff(25), p_diff(1), var_i(1), 0...]
__global__ void k_edgefeat(const float* __restrict__ u, const float* __restrict__ pos,
                           const int* __restrict__ esrc, const int* __restrict__ etgt,
                           f16* __restrict__ efeat) {
  int e = blockIdx.x * blockDim.x + threadIdx.x;
  if (e >= NE) return;
  int s = esrc[e], t = etgt[e];
  f16* d = efeat + (size_t)e * 32;
#pragma unroll
  for (int k = 0; k < 25; k++)
    d[k] = (f16)(u[(size_t)t * TWIN + k] - u[(size_t)s * TWIN + k]);
  d[25] = (f16)((pos[t * 2 + 1] - pos[s * 2 + 1]) * (1.0f / 16.0f));
  d[26] = (f16)(pos[t * 2 + 0] * 0.25f);
#pragma unroll
  for (int k = 27; k < 32; k++) d[k] = (f16)0.0f;
}

__global__ void k_cnt(const int* __restrict__ etgt, float* __restrict__ cnt) {
  int e = blockIdx.x * blockDim.x + threadIdx.x;
  if (e >= NE) return;
  atomic_add_f32(cnt + etgt[e], 1.0f);
}

// cooperative global->LDS copy of swizzled weight fragments (16B granules)
__device__ __forceinline__ void stage_weights(const f16* g, f16* s, int bytes) {
  const uint4* gs = (const uint4*)g;
  uint4* ss = (uint4*)s;
  int n4 = bytes >> 4;
  for (int i = threadIdx.x; i < n4; i += 256) ss[i] = gs[i];
}

// ---------------------------------------------------------------------------
// Encoder: h = swish(swish(nfeat @ We1 + be1) @ We2 + be2)  (1 + 4 K-chunks)
// One wave per 16-node tile; 8 waves/block; grid = NN/16/8 = 256 (exact).
// Weights staged in LDS once per block; B fragments read via ds_load.
// ---------------------------------------------------------------------------
__global__ void k_encoder(const f16* __restrict__ nfeat,
                          const f16* __restrict__ w1s, const float* __restrict__ b1,
                          const f16* __restrict__ w2s, const float* __restrict__ b2,
                          float* __restrict__ h32, f16* __restrict__ h16) {
  extern __shared__ __align__(16) char dynsmem[];
  f16* sW1 = (f16*)dynsmem;
  f16* sW2 = (f16*)(dynsmem + W1_BYTES_1);
  f16* sAct = (f16*)(dynsmem + ACT_OFF_1);
  stage_weights(w1s, sW1, W1_BYTES_1);
  stage_weights(w2s, sW2, W2_BYTES);
  __syncthreads();

  const int lane = threadIdx.x & 31;
  const int wid  = threadIdx.x >> 5;
  const int n0   = (blockIdx.x * 8 + wid) * 16;
  const int r    = lane & 15;
  const int kb2  = (lane >> 4) * 8;
  const int col  = lane & 15;
  const int mh   = (lane >> 4) * 8;
  const v16h* bw1 = (const v16h*)sW1;
  const v16h* bw2 = (const v16h*)sW2;
  const f16* rown = nfeat + (size_t)(n0 + r) * 32;

  v8f zacc = {};
  v8f acc[8];
#pragma unroll
  for (int t = 0; t < 8; t++) acc[t] = zacc;

  {
    v16h a = load_a_frag(rown, 0, kb2);
#pragma unroll
    for (int t = 0; t < 8; t++) {
      v16h b = bw1[t * 32 + lane];
      acc[t] = __builtin_amdgcn_wmma_f32_16x16x32_f16(false, a, false, b,
                                                      (short)0, acc[t], false, false);
    }
  }

  f16* lrow = sAct + wid * 16 * HD;
#pragma unroll
  for (int t = 0; t < 8; t++) {
    float bb = b1[t * 16 + col];
#pragma unroll
    for (int j = 0; j < 8; j++) {
      float x = acc[t][j] + bb;
      lrow[(mh + j) * HD + t * 16 + col] = (f16)swish_f(x);
    }
  }
  asm volatile("s_wait_dscnt 0" ::: "memory");

#pragma unroll
  for (int t = 0; t < 8; t++) acc[t] = zacc;
  const f16* lr = lrow + r * HD;
#pragma unroll
  for (int c = 0; c < 4; c++) {
    v16h a = load_a_frag(lr, c * 32, kb2);
#pragma unroll
    for (int t = 0; t < 8; t++) {
      v16h b = bw2[(c * 8 + t) * 32 + lane];
      acc[t] = __builtin_amdgcn_wmma_f32_16x16x32_f16(false, a, false, b,
                                                      (short)0, acc[t], false, false);
    }
  }

#pragma unroll
  for (int t = 0; t < 8; t++) {
    float bb = b2[t * 16 + col];
#pragma unroll
    for (int j = 0; j < 8; j++) {
      float x = acc[t][j] + bb;
      float sw = swish_f(x);
      size_t idx = (size_t)(n0 + mh + j) * HD + t * 16 + col;
      h32[idx] = sw;
      h16[idx] = (f16)sw;
    }
  }
}

// ---------------------------------------------------------------------------
// Edge message: msg = swish(swish([h_t|h_s|ef] @ Wm1 + bm1) @ Wm2 + bm2);
// atomic scatter-add into agg[tgt]. One wave per 16-edge tile; 9+4 K-chunks.
// grid = NE/16/8 = 2048 (exact). Weights staged in LDS once per block.
// ---------------------------------------------------------------------------
__global__ void k_edge(const f16* __restrict__ hx, const f16* __restrict__ efeat,
                       const int* __restrict__ esrc, const int* __restrict__ etgt,
                       const f16* __restrict__ w1s, const float* __restrict__ b1,
                       const f16* __restrict__ w2s, const float* __restrict__ b2,
                       float* __restrict__ agg) {
  extern __shared__ __align__(16) char dynsmem[];
  f16* sW1 = (f16*)dynsmem;
  f16* sW2 = (f16*)(dynsmem + W1_BYTES_9);
  f16* sAct = (f16*)(dynsmem + ACT_OFF_9);
  stage_weights(w1s, sW1, W1_BYTES_9);
  stage_weights(w2s, sW2, W2_BYTES);
  __syncthreads();

  const int lane = threadIdx.x & 31;
  const int wid  = threadIdx.x >> 5;
  const int e0   = (blockIdx.x * 8 + wid) * 16;
  const int r    = lane & 15;
  const int kb2  = (lane >> 4) * 8;
  const int col  = lane & 15;
  const int mh   = (lane >> 4) * 8;
  const int e    = e0 + r;
  const int vs = esrc[e], vt = etgt[e];
  const f16* rowt = hx + (size_t)vt * HD;
  const f16* rows = hx + (size_t)vs * HD;
  const f16* rowe = efeat + (size_t)e * 32;
  const v16h* bw1 = (const v16h*)sW1;
  const v16h* bw2 = (const v16h*)sW2;

  v8f zacc = {};
  v8f acc[8];
#pragma unroll
  for (int t = 0; t < 8; t++) acc[t] = zacc;

#pragma unroll
  for (int c = 0; c < 9; c++) {
    const f16* prow; int k0;
    if (c < 4)      { prow = rowt; k0 = c * 32; }
    else if (c < 8) { prow = rows; k0 = (c - 4) * 32; }
    else            { prow = rowe; k0 = 0; }
    v16h a = load_a_frag(prow, k0, kb2);
#pragma unroll
    for (int t = 0; t < 8; t++) {
      v16h b = bw1[(c * 8 + t) * 32 + lane];
      acc[t] = __builtin_amdgcn_wmma_f32_16x16x32_f16(false, a, false, b,
                                                      (short)0, acc[t], false, false);
    }
  }

  f16* lrow = sAct + wid * 16 * HD;
#pragma unroll
  for (int t = 0; t < 8; t++) {
    float bb = b1[t * 16 + col];
#pragma unroll
    for (int j = 0; j < 8; j++) {
      float x = acc[t][j] + bb;
      lrow[(mh + j) * HD + t * 16 + col] = (f16)swish_f(x);
    }
  }
  asm volatile("s_wait_dscnt 0" ::: "memory");

#pragma unroll
  for (int t = 0; t < 8; t++) acc[t] = zacc;
  const f16* lr = lrow + r * HD;
#pragma unroll
  for (int c = 0; c < 4; c++) {
    v16h a = load_a_frag(lr, c * 32, kb2);
#pragma unroll
    for (int t = 0; t < 8; t++) {
      v16h b = bw2[(c * 8 + t) * 32 + lane];
      acc[t] = __builtin_amdgcn_wmma_f32_16x16x32_f16(false, a, false, b,
                                                      (short)0, acc[t], false, false);
    }
  }

#pragma unroll
  for (int j = 0; j < 8; j++) {
    int ee = e0 + mh + j;
    float* dst = agg + (size_t)etgt[ee] * HD;
#pragma unroll
    for (int t = 0; t < 8; t++) {
      float x = acc[t][j] + b2[t * 16 + col];
      atomic_add_f32(dst + t * 16 + col, swish_f(x));
    }
  }
}

__global__ void k_aggnorm(const float* __restrict__ agg, const float* __restrict__ cnt,
                          f16* __restrict__ aggn) {
  int idx = blockIdx.x * blockDim.x + threadIdx.x;
  if (idx >= NN * HD) return;
  float c = cnt[idx >> 7];
  c = c < 1.0f ? 1.0f : c;
  aggn[idx] = (f16)(agg[idx] / c);
}

// ---------------------------------------------------------------------------
// Node update: h += swish(swish([h|agg|var] @ Wu1 + bu1) @ Wu2 + bu2)
// ---------------------------------------------------------------------------
__global__ void k_update(const f16* __restrict__ hx, const f16* __restrict__ aggn,
                         const f16* __restrict__ upfeat,
                         const f16* __restrict__ w1s, const float* __restrict__ b1,
                         const f16* __restrict__ w2s, const float* __restrict__ b2,
                         float* __restrict__ h32) {
  extern __shared__ __align__(16) char dynsmem[];
  f16* sW1 = (f16*)dynsmem;
  f16* sW2 = (f16*)(dynsmem + W1_BYTES_9);
  f16* sAct = (f16*)(dynsmem + ACT_OFF_9);
  stage_weights(w1s, sW1, W1_BYTES_9);
  stage_weights(w2s, sW2, W2_BYTES);
  __syncthreads();

  const int lane = threadIdx.x & 31;
  const int wid  = threadIdx.x >> 5;
  const int n0   = (blockIdx.x * 8 + wid) * 16;
  const int r    = lane & 15;
  const int kb2  = (lane >> 4) * 8;
  const int col  = lane & 15;
  const int mh   = (lane >> 4) * 8;
  const int n    = n0 + r;
  const f16* rowh = hx + (size_t)n * HD;
  const f16* rowa = aggn + (size_t)n * HD;
  const f16* rowu = upfeat + (size_t)n * 32;
  const v16h* bw1 = (const v16h*)sW1;
  const v16h* bw2 = (const v16h*)sW2;

  v8f zacc = {};
  v8f acc[8];
#pragma unroll
  for (int t = 0; t < 8; t++) acc[t] = zacc;

#pragma unroll
  for (int c = 0; c < 9; c++) {
    const f16* prow; int k0;
    if (c < 4)      { prow = rowh; k0 = c * 32; }
    else if (c < 8) { prow = rowa; k0 = (c - 4) * 32; }
    else            { prow = rowu; k0 = 0; }
    v16h a = load_a_frag(prow, k0, kb2);
#pragma unroll
    for (int t = 0; t < 8; t++) {
      v16h b = bw1[(c * 8 + t) * 32 + lane];
      acc[t] = __builtin_amdgcn_wmma_f32_16x16x32_f16(false, a, false, b,
                                                      (short)0, acc[t], false, false);
    }
  }

  f16* lrow = sAct + wid * 16 * HD;
#pragma unroll
  for (int t = 0; t < 8; t++) {
    float bb = b1[t * 16 + col];
#pragma unroll
    for (int j = 0; j < 8; j++) {
      float x = acc[t][j] + bb;
      lrow[(mh + j) * HD + t * 16 + col] = (f16)swish_f(x);
    }
  }
  asm volatile("s_wait_dscnt 0" ::: "memory");

#pragma unroll
  for (int t = 0; t < 8; t++) acc[t] = zacc;
  const f16* lr = lrow + r * HD;
#pragma unroll
  for (int c = 0; c < 4; c++) {
    v16h a = load_a_frag(lr, c * 32, kb2);
#pragma unroll
    for (int t = 0; t < 8; t++) {
      v16h b = bw2[(c * 8 + t) * 32 + lane];
      acc[t] = __builtin_amdgcn_wmma_f32_16x16x32_f16(false, a, false, b,
                                                      (short)0, acc[t], false, false);
    }
  }

#pragma unroll
  for (int t = 0; t < 8; t++) {
    float bb = b2[t * 16 + col];
#pragma unroll
    for (int j = 0; j < 8; j++) {
      float x = acc[t][j] + bb;
      size_t idx = (size_t)(n0 + mh + j) * HD + t * 16 + col;
      h32[idx] = h32[idx] + swish_f(x);   // residual; unique (node,col) per lane
    }
  }
}

// per-batch stats: 128 nodes/block (all in one batch: 4096 nodes per batch)
__global__ void k_stats(const float* __restrict__ h32, float* __restrict__ stats) {
  __shared__ float red[512];
  const int tid  = threadIdx.x;
  const int n0   = blockIdx.x * 128;
  const int b    = n0 >> 12;
  const int f    = tid & 127;
  const int half = tid >> 7;
  float s = 0.0f, q = 0.0f;
  for (int i = half; i < 128; i += 2) {
    float v = h32[(size_t)(n0 + i) * HD + f];
    s += v; q += v * v;
  }
  red[half * 128 + f] = s;
  red[256 + half * 128 + f] = q;
  __syncthreads();
  if (tid < 128) {
    float ss = red[tid] + red[128 + tid];
    float qq = red[256 + tid] + red[384 + tid];
    atomic_add_f32(stats + b * 256 + tid, ss);
    atomic_add_f32(stats + b * 256 + 128 + tid, qq);
  }
}

__global__ void k_norm(float* __restrict__ h32, const float* __restrict__ stats,
                       f16* __restrict__ h16) {
  int idx = blockIdx.x * blockDim.x + threadIdx.x;
  if (idx >= NN * HD) return;
  int f = idx & 127;
  int b = (idx >> 7) >> 12;
  float mean = stats[b * 256 + f] * (1.0f / 4096.0f);
  float msq  = stats[b * 256 + 128 + f] * (1.0f / 4096.0f);
  float inv  = rsqrtf(msq - mean * mean + 1e-5f);
  float v = (h32[idx] - mean) * inv;
  h32[idx] = v;
  h16[idx] = (f16)v;
}

// decoder: conv1d(k=16,s=3) -> swish -> conv1d(k=14) -> out; one wave per node
__global__ void k_decoder(const float* __restrict__ h32, const float* __restrict__ u,
                          const float* __restrict__ Wc1, const float* __restrict__ bc1,
                          const float* __restrict__ Wc2, const float* __restrict__ bc2,
                          float* __restrict__ out) {
  __shared__ float hb[8][HD];
  __shared__ float cb[8][8 * 38];
  const int lane = threadIdx.x & 31;
  const int wid  = threadIdx.x >> 5;
  const int n    = blockIdx.x * 8 + wid;
  const float* hr = h32 + (size_t)n * HD;
  for (int i = lane; i < HD; i += 32) hb[wid][i] = hr[i];
  asm volatile("s_wait_dscnt 0" ::: "memory");
  for (int p = lane; p < 8 * 38; p += 32) {
    int o = p / 38, j = p % 38;
    float s = bc1[o];
#pragma unroll
    for (int k = 0; k < 16; k++) s += hb[wid][3 * j + k] * Wc1[o * 16 + k];
    cb[wid][o * 38 + j] = swish_f(s);
  }
  asm volatile("s_wait_dscnt 0" ::: "memory");
  if (lane < TWIN) {
    float s = bc2[0];
#pragma unroll
    for (int o = 0; o < 8; o++)
#pragma unroll
      for (int k = 0; k < 14; k++) s += cb[wid][o * 38 + lane + k] * Wc2[o * 14 + k];
    out[(size_t)n * TWIN + lane] =
        u[(size_t)n * TWIN + 24] + (float)(lane + 1) * (4.0f / 250.0f) * s;
  }
}

// ---------------------------------------------------------------------------
extern "C" void kernel_launch(void* const* d_in, const int* in_sizes, int n_in,
                              void* d_out, int out_size, void* d_ws, size_t ws_size,
                              hipStream_t stream) {
  (void)in_sizes; (void)n_in; (void)out_size; (void)ws_size;
  const float* u   = (const float*)d_in[0];
  const float* pos = (const float*)d_in[1];
  const int*   ei  = (const int*)d_in[2];
  // d_in[3] (batch) derived analytically: batch = node >> 12
  const float* We1 = (const float*)d_in[4];
  const float* be1 = (const float*)d_in[5];
  const float* We2 = (const float*)d_in[6];
  const float* be2 = (const float*)d_in[7];
  const float* Wm1 = (const float*)d_in[8];
  const float* bm1 = (const float*)d_in[9];
  const float* Wm2 = (const float*)d_in[10];
  const float* bm2 = (const float*)d_in[11];
  const float* Wu1 = (const float*)d_in[12];
  const float* bu1 = (const float*)d_in[13];
  const float* Wu2 = (const float*)d_in[14];
  const float* bu2 = (const float*)d_in[15];
  const float* Wc1 = (const float*)d_in[16];
  const float* bc1 = (const float*)d_in[17];
  const float* Wc2 = (const float*)d_in[18];
  const float* bc2 = (const float*)d_in[19];
  const int* esrc = ei;
  const int* etgt = ei + NE;
  float* out = (float*)d_out;

  uint8_t* wsb = (uint8_t*)d_ws;
  size_t off = 0;
  auto carve = [&](size_t bytes) -> void* {
    off = (off + 255) & ~(size_t)255;
    void* p = wsb + off;
    off += bytes;
    return p;
  };
  float* h32   = (float*)carve((size_t)NN * HD * 4);
  f16*   h16   = (f16*)  carve((size_t)NN * HD * 2);
  float* agg   = (float*)carve((size_t)NN * HD * 4);
  f16*   aggn  = (f16*)  carve((size_t)NN * HD * 2);
  f16*   efeat = (f16*)  carve((size_t)NE * 32 * 2);
  f16*   nfeat = (f16*)  carve((size_t)NN * 32 * 2);
  f16*   upft  = (f16*)  carve((size_t)NN * 32 * 2);
  float* cnt   = (float*)carve((size_t)NN * 4);
  float* stats = (float*)carve((size_t)NBATCH * 256 * 4);
  // swizzled weights: fragment = 512 f16; per chunk 8 fragments = 4096 f16
  f16* We1s = (f16*)carve((size_t)1 * 4096 * 2);
  f16* We2s = (f16*)carve((size_t)4 * 4096 * 2);
  f16* Wm1s = (f16*)carve((size_t)NLAY * 9 * 4096 * 2);
  f16* Wm2s = (f16*)carve((size_t)NLAY * 4 * 4096 * 2);
  f16* Wu1s = (f16*)carve((size_t)NLAY * 9 * 4096 * 2);
  f16* Wu2s = (f16*)carve((size_t)NLAY * 4 * 4096 * 2);

  k_swizzle<<<1, 256, 0, stream>>>(We1, 27, 1, We1s);
  k_swizzle<<<4, 256, 0, stream>>>(We2, 128, 4, We2s);
  for (int l = 0; l < NLAY; l++) {
    k_swizzle<<<9, 256, 0, stream>>>(Wm1 + (size_t)l * 283 * HD, 283, 9, Wm1s + (size_t)l * 9 * 4096);
    k_swizzle<<<4, 256, 0, stream>>>(Wm2 + (size_t)l * 128 * HD, 128, 4, Wm2s + (size_t)l * 4 * 4096);
    k_swizzle<<<9, 256, 0, stream>>>(Wu1 + (size_t)l * 257 * HD, 257, 9, Wu1s + (size_t)l * 9 * 4096);
    k_swizzle<<<4, 256, 0, stream>>>(Wu2 + (size_t)l * 128 * HD, 128, 4, Wu2s + (size_t)l * 4 * 4096);
  }

  k_nodefeat<<<NN / 256, 256, 0, stream>>>(u, pos, nfeat, upft);
  k_edgefeat<<<NE / 256, 256, 0, stream>>>(u, pos, esrc, etgt, efeat);
  k_zero<<<NN / 256, 256, 0, stream>>>(cnt, NN);
  k_cnt<<<NE / 256, 256, 0, stream>>>(etgt, cnt);
  k_encoder<<<NN / 128, 256, SMEM_1, stream>>>(nfeat, We1s, be1, We2s, be2, h32, h16);

  for (int l = 0; l < NLAY; l++) {
    k_zero<<<NN * HD / 256, 256, 0, stream>>>(agg, NN * HD);
    k_edge<<<NE / 128, 256, SMEM_9, stream>>>(h16, efeat, esrc, etgt,
                                              Wm1s + (size_t)l * 9 * 4096, bm1 + l * HD,
                                              Wm2s + (size_t)l * 4 * 4096, bm2 + l * HD, agg);
    k_aggnorm<<<NN * HD / 256, 256, 0, stream>>>(agg, cnt, aggn);
    k_update<<<NN / 128, 256, SMEM_9, stream>>>(h16, aggn, upft,
                                                Wu1s + (size_t)l * 9 * 4096, bu1 + l * HD,
                                                Wu2s + (size_t)l * 4 * 4096, bu2 + l * HD, h32);
    k_zero<<<(NBATCH * 256 + 255) / 256, 256, 0, stream>>>(stats, NBATCH * 256);
    k_stats<<<NN / 128, 256, 0, stream>>>(h32, stats);
    k_norm<<<NN * HD / 256, 256, 0, stream>>>(h32, stats, h16);
  }

  k_decoder<<<NN / 8, 256, 0, stream>>>(h32, u, Wc1, bc1, Wc2, bc2, out);
}
